// AttentionLayer_83313775607828
// MI455X (gfx1250) — matrix operand order repaired and checked by reading, not compiled, not run on hardware
//
#include <hip/hip_runtime.h>

typedef __attribute__((ext_vector_type(16))) _Float16 v16h;
typedef __attribute__((ext_vector_type(8)))  _Float16 v8h;
typedef __attribute__((ext_vector_type(8)))  float    v8f;
typedef __attribute__((ext_vector_type(4)))  float    v4f;

#define NB   8
#define TQ   128
#define TV   256
#define DD   512
#define UU   256

#define W1V_STRIDE 264   // 256 + 8 f16 pad: rows stay 16B aligned, banks de-padded
#define P_STRIDE   264

__device__ __forceinline__ float fast_exp2(float x) {
#if __has_builtin(__builtin_amdgcn_exp2f)
  return __builtin_amdgcn_exp2f(x);
#else
  return exp2f(x);
#endif
}

__device__ __forceinline__ float fast_rcp(float x) {
#if __has_builtin(__builtin_amdgcn_rcpf)
  return __builtin_amdgcn_rcpf(x);
#else
  return 1.0f / x;
#endif
}

__device__ __forceinline__ float fast_tanh(float x) {
#if __has_builtin(__builtin_amdgcn_tanhf)
  return __builtin_amdgcn_tanhf(x);          // v_tanh_f32 on gfx1250
#elif __has_builtin(__builtin_amdgcn_tanh_f32)
  return __builtin_amdgcn_tanh_f32(x);
#else
  // tanh(x) = 1 - 2/(exp(2x)+1);  2*log2(e) = 2.885390081777927
  float e = fast_exp2(x * 2.885390081777927f);
  return 1.0f - 2.0f * fast_rcp(e + 1.0f);
#endif
}

__device__ __forceinline__ v8f wmma16(v16h a, v16h b, v8f c) {
  return __builtin_amdgcn_wmma_f32_16x16x32_f16(false, a, false, b,
                                                (short)0, c, false, false);
}

// A fragment (16x32 f16) from a global f32 row: K runs {akb..akb+7, akb+16..akb+23}
// are two 32B-aligned contiguous blocks -> 4x b128 loads.
__device__ __forceinline__ v16h load_a_row_f32(const float* __restrict__ row,
                                               int k0, int lhi) {
  const int akb = lhi ? 8 : 0;
  const v4f* p0 = (const v4f*)(row + k0 + akb);
  const v4f* p1 = (const v4f*)(row + k0 + akb + 16);
  v4f x0 = p0[0], x1 = p0[1], x2 = p1[0], x3 = p1[1];
  v16h a;
  #pragma unroll
  for (int i = 0; i < 4; ++i) {
    a[i]      = (_Float16)x0[i];
    a[4 + i]  = (_Float16)x1[i];
    a[8 + i]  = (_Float16)x2[i];
    a[12 + i] = (_Float16)x3[i];
  }
  return a;
}

// B fragment (32x16 f16): lane = column, K = {bkb..bkb+7, bkb+16..bkb+23} (row-strided, scalar loads).
__device__ __forceinline__ v16h load_b_col_f32(const float* __restrict__ base,
                                               int ld, int k0, int col, int lhi) {
  const int bkb = lhi ? 8 : 0;
  v16h b;
  #pragma unroll
  for (int j = 0; j < 16; ++j)
    b[j] = (_Float16)base[(size_t)(k0 + bkb + j + (j >= 8 ? 8 : 0)) * ld + col];
  return b;
}

// A fragment from an LDS f16 row: two contiguous 8xf16 (16B) runs -> 2x ds_load_b128.
__device__ __forceinline__ v16h load_a_row_lds(const _Float16* row, int k0, int lhi) {
  const int akb = lhi ? 8 : 0;
  v8h x0 = *(const v8h*)(row + k0 + akb);
  v8h x1 = *(const v8h*)(row + k0 + akb + 16);
  v16h a;
  #pragma unroll
  for (int i = 0; i < 8; ++i) { a[i] = x0[i]; a[8 + i] = x1[i]; }
  return a;
}

__global__ __launch_bounds__(256) void attn_fused_kernel(
    const float* __restrict__ query,   // [NB, TQ, DD]
    const float* __restrict__ values,  // [NB, TV, DD]
    const float* __restrict__ W1,      // [DD, UU]
    const float* __restrict__ b1,      // [UU]
    const float* __restrict__ W2,      // [DD, UU]
    const float* __restrict__ b2,      // [UU]
    const float* __restrict__ Vv,      // [UU, 1]
    const float* __restrict__ bv,      // [1]
    float* __restrict__ out)           // [NB, TQ, DD]
{
  // ---- LDS (~176 KB of the 320 KB/WGP) ----
  __shared__ __align__(16) _Float16 s_w1v[TV][W1V_STRIDE];  // 135 KB
  __shared__ __align__(16) float    s_w2q[16][UU];          // 16 KB
  __shared__ __align__(16) float    s_score[16][TV];        // 16 KB
  __shared__ __align__(16) _Float16 s_p[16][P_STRIDE];      // 8.3 KB
  __shared__ __align__(16) float    s_V[UU];                // 1 KB
  __shared__ float s_redmax[16][17];
  __shared__ float s_redsum[16][17];

  const int bidx = blockIdx.y;          // batch
  const int q0   = blockIdx.x * 16;     // query tile base
  const int tid  = threadIdx.x;
  const int wave = tid >> 5;
  const int lane = tid & 31;
  const int lm   = lane & 15;           // M/N within tile
  const int lhi  = lane >> 4;           // lane half selects K groups

  const float* vbase = values + (size_t)bidx * TV * DD;
  const float* qbase = query  + ((size_t)bidx * TQ + q0) * DD;

  // Warm L2/WGP$ for values[b] (global_prefetch_b8); one row start per thread.
  __builtin_prefetch(vbase + (size_t)tid * DD, 0, 3);

  // ====== Phase 0a: w1v = values@W1 + b1  (WMMA; B-fragment reused 4x) ======
  for (int ut = wave * 2; ut < wave * 2 + 2; ++ut) {   // 16 u-tiles / 8 waves
    const int u0 = ut * 16;
    const float bias = b1[u0 + lm];
    for (int vt0 = 0; vt0 < 16; vt0 += 4) {            // 4 v-tiles per pass
      v8f c0 = {}, c1 = {}, c2 = {}, c3 = {};
      for (int k0 = 0; k0 < DD; k0 += 32) {
        v16h bm = load_b_col_f32(W1, UU, k0, u0 + lm, lhi);
        v16h a0 = load_a_row_f32(vbase + (size_t)((vt0 + 0) * 16 + lm) * DD, k0, lhi);
        c0 = wmma16(a0, bm, c0);
        v16h a1 = load_a_row_f32(vbase + (size_t)((vt0 + 1) * 16 + lm) * DD, k0, lhi);
        c1 = wmma16(a1, bm, c1);
        v16h a2 = load_a_row_f32(vbase + (size_t)((vt0 + 2) * 16 + lm) * DD, k0, lhi);
        c2 = wmma16(a2, bm, c2);
        v16h a3 = load_a_row_f32(vbase + (size_t)((vt0 + 3) * 16 + lm) * DD, k0, lhi);
        c3 = wmma16(a3, bm, c3);
      }
      #pragma unroll
      for (int r = 0; r < 8; ++r) {
        const int rr = r + (lhi ? 8 : 0);
        s_w1v[(vt0 + 0) * 16 + rr][u0 + lm] = (_Float16)(c0[r] + bias);
        s_w1v[(vt0 + 1) * 16 + rr][u0 + lm] = (_Float16)(c1[r] + bias);
        s_w1v[(vt0 + 2) * 16 + rr][u0 + lm] = (_Float16)(c2[r] + bias);
        s_w1v[(vt0 + 3) * 16 + rr][u0 + lm] = (_Float16)(c3[r] + bias);
      }
    }
  }

  // ====== Phase 0b: w2q tile = query@W2 + b2  (A-fragment reused 2x) =======
  {
    const int u0a = (wave * 2) * 16;
    const int u0b = u0a + 16;
    v8f c0 = {}, c1 = {};
    for (int k0 = 0; k0 < DD; k0 += 32) {
      v16h a = load_a_row_f32(qbase + (size_t)lm * DD, k0, lhi);
      v16h bm0 = load_b_col_f32(W2, UU, k0, u0a + lm, lhi);
      c0 = wmma16(a, bm0, c0);
      v16h bm1 = load_b_col_f32(W2, UU, k0, u0b + lm, lhi);
      c1 = wmma16(a, bm1, c1);
    }
    const float biasa = b2[u0a + lm];
    const float biasb = b2[u0b + lm];
    #pragma unroll
    for (int r = 0; r < 8; ++r) {
      const int rr = r + (lhi ? 8 : 0);
      s_w2q[rr][u0a + lm] = c0[r] + biasa;
      s_w2q[rr][u0b + lm] = c1[r] + biasb;
    }
  }

  // Phase 0c: stage V
  s_V[tid] = Vv[tid];                  // blockDim == UU == 256
  __syncthreads();

  // ===== Phase 1: score[q,v] = sum_u V[u]*tanh(w1v[v,u]+w2q[q,u]) ==========
  const int q  = tid >> 4;             // 0..15
  const int vl = tid & 15;
  const float bvv = bv[0];
  for (int v = vl; v < TV; v += 16) {
    float acc = 0.0f;
    const v8h* w1p = (const v8h*)&s_w1v[v][0];   // 1x b128 per 8 u
    const v4f* w2p = (const v4f*)&s_w2q[q][0];   // broadcast across 16 lanes
    const v4f* vp  = (const v4f*)&s_V[0];        // broadcast across all lanes
    #pragma unroll 2
    for (int u8 = 0; u8 < UU / 8; ++u8) {
      v8h h  = w1p[u8];
      v4f wa = w2p[2 * u8], wb = w2p[2 * u8 + 1];
      v4f va = vp[2 * u8],  vb = vp[2 * u8 + 1];
      #pragma unroll
      for (int i = 0; i < 4; ++i) {
        acc = fmaf(va[i], fast_tanh((float)h[i]     + wa[i]), acc);
        acc = fmaf(vb[i], fast_tanh((float)h[4 + i] + wb[i]), acc);
      }
    }
    s_score[q][v] = acc + bvv;
  }
  __syncthreads();

  // ================= Phase 2: softmax over v (row-wise) ====================
  {
    const int seg = vl;                // 16 contiguous v per thread
    float mx = -3.0e38f;
    #pragma unroll
    for (int i = 0; i < 16; ++i) mx = fmaxf(mx, s_score[q][seg * 16 + i]);
    s_redmax[q][seg] = mx;
    __syncthreads();
    if (seg == 0) {
      float mm = s_redmax[q][0];
      #pragma unroll
      for (int i = 1; i < 16; ++i) mm = fmaxf(mm, s_redmax[q][i]);
      s_redmax[q][0] = mm;
    }
    __syncthreads();
    const float rowmax = s_redmax[q][0];
    const float L2E = 1.4426950408889634f;
    float ssum = 0.0f;
    #pragma unroll
    for (int i = 0; i < 16; ++i) {
      int v = seg * 16 + i;
      float e = fast_exp2((s_score[q][v] - rowmax) * L2E);
      s_score[q][v] = e;
      ssum += e;
    }
    s_redsum[q][seg] = ssum;
    __syncthreads();
    if (seg == 0) {
      float t = 0.0f;
      #pragma unroll
      for (int i = 0; i < 16; ++i) t += s_redsum[q][i];
      s_redsum[q][0] = t;
    }
    __syncthreads();
    const float rinv = fast_rcp(s_redsum[q][0]);
    #pragma unroll
    for (int i = 0; i < 16; ++i) {
      int v = seg * 16 + i;
      s_p[q][v] = (_Float16)(s_score[q][v] * rinv);
    }
  }
  __syncthreads();

  // ====== Phase 3: context = P[16xTV] @ values[TVxDD] (A reused 4x) ========
  {
    const int n0 = wave * 64;          // 4 consecutive 16-col tiles per wave
    const _Float16* prow = &s_p[lm][0];
    v8f c0 = {}, c1 = {}, c2 = {}, c3 = {};
    for (int k0 = 0; k0 < TV; k0 += 32) {
      v16h a  = load_a_row_lds(prow, k0, lhi);
      v16h b0 = load_b_col_f32(vbase, DD, k0, n0 +  0 + lm, lhi);
      c0 = wmma16(a, b0, c0);
      v16h b1f = load_b_col_f32(vbase, DD, k0, n0 + 16 + lm, lhi);
      c1 = wmma16(a, b1f, c1);
      v16h b2f = load_b_col_f32(vbase, DD, k0, n0 + 32 + lm, lhi);
      c2 = wmma16(a, b2f, c2);
      v16h b3f = load_b_col_f32(vbase, DD, k0, n0 + 48 + lm, lhi);
      c3 = wmma16(a, b3f, c3);
    }
    float* obase = out + ((size_t)bidx * TQ + q0) * DD;
    #pragma unroll
    for (int r = 0; r < 8; ++r) {
      const int rr = r + (lhi ? 8 : 0);
      obase[(size_t)rr * DD + n0 +  0 + lm] = c0[r];
      obase[(size_t)rr * DD + n0 + 16 + lm] = c1[r];
      obase[(size_t)rr * DD + n0 + 32 + lm] = c2[r];
      obase[(size_t)rr * DD + n0 + 48 + lm] = c3[r];
    }
  }
}

extern "C" void kernel_launch(void* const* d_in, const int* in_sizes, int n_in,
                              void* d_out, int out_size, void* d_ws, size_t ws_size,
                              hipStream_t stream) {
  (void)in_sizes; (void)n_in; (void)d_ws; (void)ws_size; (void)out_size;
  const float* query  = (const float*)d_in[0];
  const float* values = (const float*)d_in[1];
  const float* W1     = (const float*)d_in[2];
  const float* b1     = (const float*)d_in[3];
  const float* W2     = (const float*)d_in[4];
  const float* b2     = (const float*)d_in[5];
  const float* Vv     = (const float*)d_in[6];
  const float* bv     = (const float*)d_in[7];
  float* out = (float*)d_out;

  dim3 grid(TQ / 16, NB, 1);   // 8 query tiles x 8 batches = 64 workgroups
  dim3 block(256, 1, 1);       // 8 wave32
  attn_fused_kernel<<<grid, block, 0, stream>>>(query, values, W1, b1,
                                                W2, b2, Vv, bv, out);
}
